// CrossStreamFusionAttention_8108898255544
// MI455X (gfx1250) — matrix operand order repaired and checked by reading, not compiled, not run on hardware
//
#include <hip/hip_runtime.h>
#include <hip/hip_bf16.h>

typedef __attribute__((ext_vector_type(16))) __bf16 v16bf;
typedef __attribute__((ext_vector_type(8)))  __bf16 v8bf;
typedef __attribute__((ext_vector_type(8)))  float  v8f;

#define NTOK 49152
#define CDIM 512
#define HNUM 8
#define KPAT 48
#define DDIM 64
#define PNUM 1024
#define ATT_SCALE 0.125f
#define LN_EPS 1e-5f

// ---------------- async global->LDS helpers (CDNA5 ASYNCcnt path) ----------------
static __device__ inline unsigned lds_off_of(const void* p) {
  // addrspacecast generic -> local gives the 32-bit LDS offset the async op needs
  return (unsigned)(size_t)(__attribute__((address_space(3))) const void*)p;
}

static __device__ inline void async_copy_b128(unsigned ldsOff, const void* gptr) {
  unsigned long long ga = (unsigned long long)(size_t)gptr;
  asm volatile("global_load_async_to_lds_b128 %0, %1, off"
               :: "v"(ldsOff), "v"(ga)
               : "memory");
}

static __device__ inline void wait_async() {
  asm volatile("s_wait_asynccnt 0x0" ::: "memory");
}

// ---------------- WMMA fragment loaders (CDNA5 16x16x32 bf16 layouts) ----------------
// A (16x32): lanes 0-15 row M=lane, K {0..7} in v0-3 and {16..23} in v4-7;
//            lanes 16-31 row M=lane-16, K {8..15} and {24..31}.
static __device__ inline v16bf load_a_frag(const __bf16* s, int stride, int rowBase,
                                           int kBase, int lane) {
  int r  = rowBase + (lane & 15);
  int ks = kBase + ((lane & 16) ? 8 : 0);
  const __bf16* p = s + r * stride + ks;
  v8bf lo = *(const v8bf*)p;          // K = ks .. ks+7
  v8bf hi = *(const v8bf*)(p + 16);   // K = ks+16 .. ks+23
  v16bf o;
#pragma unroll
  for (int i = 0; i < 8; ++i) { o[i] = lo[i]; o[i + 8] = hi[i]; }
  return o;
}

// B (32x16): lanes 0-15 col N=lane, K {0..7} and {16..23}; lanes 16-31 col N=lane-16, K {8..15} and {24..31}.
// s is B transposed in LDS: row = N, col = K.
static __device__ inline v16bf load_b_frag(const __bf16* s, int stride, int nBase,
                                           int kBase, int lane) {
  int n  = nBase + (lane & 15);
  int ko = kBase + ((lane & 16) ? 8 : 0);
  const __bf16* p = s + n * stride + ko;
  v8bf lo = *(const v8bf*)p;
  v8bf hi = *(const v8bf*)(p + 16);
  v16bf o;
#pragma unroll
  for (int i = 0; i < 8; ++i) { o[i] = lo[i]; o[i + 8] = hi[i]; }
  return o;
}

// ---------------- LayerNorm + cast to bf16 (one wave per row, C=512) ----------------
__global__ __launch_bounds__(256)
void ln_cast_kernel(const float* __restrict__ x, const float* __restrict__ g,
                    const float* __restrict__ b, __bf16* __restrict__ y) {
  int wid = threadIdx.x >> 5, lane = threadIdx.x & 31;
  size_t row = (size_t)blockIdx.x * 8 + wid;
  const float* xr = x + row * CDIM;
  float v[16]; float s = 0.f;
#pragma unroll
  for (int i = 0; i < 16; ++i) { v[i] = xr[lane + 32 * i]; s += v[i]; }
#pragma unroll
  for (int o = 16; o >= 1; o >>= 1) s += __shfl_xor(s, o);
  float mean = s * (1.f / CDIM);
  float q = 0.f;
#pragma unroll
  for (int i = 0; i < 16; ++i) { float d = v[i] - mean; q += d * d; }
#pragma unroll
  for (int o = 16; o >= 1; o >>= 1) q += __shfl_xor(q, o);
  float rstd = rsqrtf(q * (1.f / CDIM) + LN_EPS);
#pragma unroll
  for (int i = 0; i < 16; ++i) {
    int c = lane + 32 * i;
    y[row * CDIM + c] = (__bf16)((v[i] - mean) * rstd * g[c] + b[c]);
  }
}

// ---------------- f32 -> bf16 cast ----------------
__global__ void cast_f32_bf16(const float* __restrict__ x, __bf16* __restrict__ y, int n) {
  int i = blockIdx.x * 256 + threadIdx.x;
  if (i < n) y[i] = (__bf16)x[i];
}

// ---------------- qcat = [q_s | q_b] ----------------
__global__ void concat_q_kernel(const __bf16* __restrict__ qs, const __bf16* __restrict__ qb,
                                __bf16* __restrict__ qc) {
  int i = blockIdx.x * 256 + threadIdx.x;   // over NTOK*CDIM, exact multiple
  int r = i / CDIM, c = i % CDIM;
  qc[(size_t)r * 1024 + c]       = qs[(size_t)r * 1536 + c];
  qc[(size_t)r * 1024 + 512 + c] = qb[(size_t)r * 1536 + c];
}

// ---------------- bf16 WMMA GEMM: C = act(A @ B + bias), fused scatter/residual ----------------
#define BM 128
#define BN 128
#define BK 32
#define LSTR 40   // LDS row stride (bf16 elems), multiple of 8, padded vs 32

template <bool OUT_BF16>
__global__ __launch_bounds__(256)
void gemm_bf16_kernel(const __bf16* __restrict__ A, int lda,
                      const __bf16* __restrict__ B, int ldb,
                      const float* __restrict__ bias,
                      void* __restrict__ Cout, int ldc,
                      int M, int Nn, int Kd, int do_gelu,
                      const int* __restrict__ scatter,
                      const float* __restrict__ residual, int ldr) {
  __shared__ __bf16 sA[BM * LSTR];
  __shared__ __bf16 sB[BN * LSTR];   // B transposed: [n][k]

  const int tid = threadIdx.x;
  const int lane = tid & 31;
  const int wid  = tid >> 5;
  const int waveM = wid & 3;      // 4 waves in M
  const int waveN = wid >> 2;     // 2 waves in N
  const int m0 = blockIdx.y * BM;
  const int n0 = blockIdx.x * BN;

  const unsigned sA_base = lds_off_of(sA);

  v8f acc[2][4];
#pragma unroll
  for (int i = 0; i < 2; ++i)
#pragma unroll
    for (int j = 0; j < 4; ++j) acc[i][j] = {};

  const int ktiles = Kd / BK;
  for (int kt = 0; kt < ktiles; ++kt) {
    const int k0 = kt * BK;
    // ---- stage A tile 128x32 via async global->LDS (16B per lane per op) ----
    {
      int r  = tid >> 1;
      int cc = (tid & 1) * 16;
      const __bf16* g = A + (size_t)(m0 + r) * lda + k0 + cc;
      if (kt + 1 < ktiles) __builtin_prefetch(g + BK, 0, 0);
      unsigned lo = sA_base + (unsigned)(r * LSTR + cc) * 2u;
      async_copy_b128(lo, g);
      async_copy_b128(lo + 16u, g + 8);
    }
    // ---- stage B tile 32x128 transposed into sB[n][k] ----
#pragma unroll
    for (int j = 0; j < 2; ++j) {
      int linear = tid + 256 * j;
      int kk = linear >> 4;
      int nc = (linear & 15) * 8;
      const __bf16* g = B + (size_t)(k0 + kk) * ldb + n0 + nc;
      uint4 vv = *(const uint4*)g;
      const __bf16* e = (const __bf16*)&vv;
#pragma unroll
      for (int q = 0; q < 8; ++q) sB[(nc + q) * LSTR + kk] = e[q];
    }
    wait_async();
    __syncthreads();

    v16bf afr[2];
#pragma unroll
    for (int mi = 0; mi < 2; ++mi)
      afr[mi] = load_a_frag(sA, LSTR, waveM * 32 + mi * 16, 0, lane);
#pragma unroll
    for (int ni = 0; ni < 4; ++ni) {
      v16bf bfr = load_b_frag(sB, LSTR, waveN * 64 + ni * 16, 0, lane);
#pragma unroll
      for (int mi = 0; mi < 2; ++mi)
        acc[mi][ni] = __builtin_amdgcn_wmma_f32_16x16x32_bf16(
            false, afr[mi], false, bfr, (short)0, acc[mi][ni], false, false);
    }
    __syncthreads();
  }

  // ---- epilogue: bias, gelu, scatter, residual ----
  const int colBase = n0 + waveN * 64;
#pragma unroll
  for (int mi = 0; mi < 2; ++mi) {
#pragma unroll
    for (int ni = 0; ni < 4; ++ni) {
      int gn = colBase + ni * 16 + (lane & 15);
      float bv = bias ? bias[gn] : 0.f;
#pragma unroll
      for (int v = 0; v < 8; ++v) {
        int gm = m0 + waveM * 32 + mi * 16 + v + ((lane & 16) ? 8 : 0);
        float val = acc[mi][ni][v] + bv;
        if (do_gelu) val = 0.5f * val * (1.f + erff(val * 0.70710678118f));
        int row = scatter ? scatter[gm] : gm;
        if (OUT_BF16) {
          ((__bf16*)Cout)[(size_t)row * ldc + gn] = (__bf16)val;
        } else {
          float r = residual ? residual[(size_t)row * ldr + gn] : 0.f;
          ((float*)Cout)[(size_t)row * ldc + gn] = val + r;
        }
      }
    }
  }
}

// ---------------- patch attention: one wave per (patch, head) ----------------
#define ATS 72   // LDS row stride, multiple of 8
__global__ __launch_bounds__(64)
void attn_kernel(const __bf16* __restrict__ qf, const __bf16* __restrict__ qkv,
                 __bf16* __restrict__ out) {
  __shared__ __bf16 smem[2][2][64 * ATS];
  int wid = threadIdx.x >> 5, lane = threadIdx.x & 31;
  int p = blockIdx.x;
  int h = blockIdx.y * 2 + wid;
  __bf16* b0 = smem[wid][0];      // Q, then softmax(A)
  __bf16* b1 = smem[wid][1];      // K, then V^T
  size_t rb = (size_t)p * KPAT;

  const unsigned b0_base = lds_off_of(b0);
  const unsigned b1_base = lds_off_of(b1);

  // stage Q (48x64) and K (48x64) via async global->LDS
#pragma unroll
  for (int i = 0; i < 12; ++i) {
    int idx = lane + 32 * i;
    int m = idx >> 3;
    int c = (idx & 7) * 8;
    unsigned loff = (unsigned)(m * ATS + c) * 2u;
    async_copy_b128(b0_base + loff, qf  + (rb + m) * CDIM + h * DDIM + c);
    async_copy_b128(b1_base + loff, qkv + (rb + m) * 1536 + CDIM + h * DDIM + c);
  }
  wait_async();
  __syncthreads();

  // S = Q K^T  (48x48 over K=64; b1 holds K row-major == B-transposed)
  v8f accS[3][3];
#pragma unroll
  for (int i = 0; i < 3; ++i)
#pragma unroll
    for (int j = 0; j < 3; ++j) accS[i][j] = {};
#pragma unroll
  for (int kj = 0; kj < 2; ++kj) {
    v16bf a[3];
#pragma unroll
    for (int mi = 0; mi < 3; ++mi) a[mi] = load_a_frag(b0, ATS, mi * 16, kj * 32, lane);
#pragma unroll
    for (int nj = 0; nj < 3; ++nj) {
      v16bf bf = load_b_frag(b1, ATS, nj * 16, kj * 32, lane);
#pragma unroll
      for (int mi = 0; mi < 3; ++mi)
        accS[mi][nj] = __builtin_amdgcn_wmma_f32_16x16x32_bf16(
            false, a[mi], false, bf, (short)0, accS[mi][nj], false, false);
    }
  }
  __syncthreads();

  // softmax per row (row m lives across one 16-lane half); write A to b0, K-padded to 64
#pragma unroll
  for (int mi = 0; mi < 3; ++mi) {
#pragma unroll
    for (int v = 0; v < 8; ++v) {
      float x0 = accS[mi][0][v] * ATT_SCALE;
      float x1 = accS[mi][1][v] * ATT_SCALE;
      float x2 = accS[mi][2][v] * ATT_SCALE;
      float mx = fmaxf(x0, fmaxf(x1, x2));
#pragma unroll
      for (int o = 8; o >= 1; o >>= 1) mx = fmaxf(mx, __shfl_xor(mx, o));
      float e0 = __expf(x0 - mx), e1 = __expf(x1 - mx), e2 = __expf(x2 - mx);
      float s = e0 + e1 + e2;
#pragma unroll
      for (int o = 8; o >= 1; o >>= 1) s += __shfl_xor(s, o);
      float inv = 1.f / s;
      int m  = mi * 16 + v + ((lane & 16) ? 8 : 0);
      int nc = lane & 15;
      b0[m * ATS + nc]      = (__bf16)(e0 * inv);
      b0[m * ATS + 16 + nc] = (__bf16)(e1 * inv);
      b0[m * ATS + 32 + nc] = (__bf16)(e2 * inv);
      b0[m * ATS + 48 + nc] = (__bf16)0.f;     // zero-pad K 48..63
    }
  }

  // stage V^T into b1: b1[d][k] = V[k][d], K-padded to 64
#pragma unroll
  for (int dd = 0; dd < 2; ++dd) {
    int d = lane + 32 * dd;
    for (int k = 0; k < KPAT; ++k)
      b1[d * ATS + k] = qkv[(rb + k) * 1536 + 2 * CDIM + h * DDIM + d];
#pragma unroll
    for (int k = KPAT; k < 64; ++k) b1[d * ATS + k] = (__bf16)0.f;
  }
  __syncthreads();

  // O = A V  (48x64 over K=48 padded to 64)
  v8f accO[3][4];
#pragma unroll
  for (int i = 0; i < 3; ++i)
#pragma unroll
    for (int j = 0; j < 4; ++j) accO[i][j] = {};
#pragma unroll
  for (int kj = 0; kj < 2; ++kj) {
    v16bf a[3];
#pragma unroll
    for (int mi = 0; mi < 3; ++mi) a[mi] = load_a_frag(b0, ATS, mi * 16, kj * 32, lane);
#pragma unroll
    for (int nj = 0; nj < 4; ++nj) {
      v16bf bf = load_b_frag(b1, ATS, nj * 16, kj * 32, lane);
#pragma unroll
      for (int mi = 0; mi < 3; ++mi)
        accO[mi][nj] = __builtin_amdgcn_wmma_f32_16x16x32_bf16(
            false, a[mi], false, bf, (short)0, accO[mi][nj], false, false);
    }
  }

  // store O to attn buffer (sorted domain), bf16
#pragma unroll
  for (int mi = 0; mi < 3; ++mi)
#pragma unroll
    for (int nj = 0; nj < 4; ++nj) {
      int gc = h * DDIM + nj * 16 + (lane & 15);
#pragma unroll
      for (int v = 0; v < 8; ++v) {
        int m = mi * 16 + v + ((lane & 16) ? 8 : 0);
        out[(rb + m) * CDIM + gc] = (__bf16)accO[mi][nj][v];
      }
    }
}

// ---------------- host launch ----------------
extern "C" void kernel_launch(void* const* d_in, const int* in_sizes, int n_in,
                              void* d_out, int out_size, void* d_ws, size_t ws_size,
                              hipStream_t stream) {
  const float* sem     = (const float*)d_in[0];
  const float* bnd     = (const float*)d_in[1];
  const int*   order   = (const int*)d_in[2];
  const int*   inverse = (const int*)d_in[3];
  const float* g_sem   = (const float*)d_in[4];
  const float* b_sem   = (const float*)d_in[5];
  const float* g_bnd   = (const float*)d_in[6];
  const float* b_bnd   = (const float*)d_in[7];
  const float* w_qkv_s = (const float*)d_in[8];
  const float* b_qkv_s = (const float*)d_in[9];
  const float* w_qkv_b = (const float*)d_in[10];
  const float* b_qkv_b = (const float*)d_in[11];
  const float* w_f1    = (const float*)d_in[12];
  const float* b_f1    = (const float*)d_in[13];
  const float* w_f2    = (const float*)d_in[14];
  const float* b_f2    = (const float*)d_in[15];
  const float* w_os    = (const float*)d_in[16];
  const float* b_os    = (const float*)d_in[17];
  const float* w_ob    = (const float*)d_in[18];
  const float* b_ob    = (const float*)d_in[19];

  char* ws = (char*)d_ws;
  size_t off = 0;
  auto alloc = [&](size_t bytes) -> char* {
    char* p = ws + off; off += (bytes + 255) & ~(size_t)255; return p;
  };
  const size_t N = NTOK;
  __bf16* QKVS = (__bf16*)alloc(N * 1536 * 2);   // sorted qkv, sem
  __bf16* QKVB = (__bf16*)alloc(N * 1536 * 2);   // sorted qkv, bnd
  __bf16* XLN  = (__bf16*)alloc(N * 512 * 2);    // LN output (reused per stream)
  __bf16* QC   = (__bf16*)alloc(N * 1024 * 2);   // qcat; later attn_s | attn_b
  __bf16* H1   = (__bf16*)alloc(N * 1024 * 2);   // gelu hidden
  __bf16* QF   = (__bf16*)alloc(N * 512 * 2);    // fused q
  __bf16* Wqs  = (__bf16*)alloc(512 * 1536 * 2);
  __bf16* Wqb  = (__bf16*)alloc(512 * 1536 * 2);
  __bf16* Wf1  = (__bf16*)alloc(1024 * 1024 * 2);
  __bf16* Wf2  = (__bf16*)alloc(1024 * 512 * 2);
  __bf16* Wos  = (__bf16*)alloc(512 * 512 * 2);
  __bf16* Wob  = (__bf16*)alloc(512 * 512 * 2);

  auto castw = [&](const float* src, __bf16* dst, int n) {
    cast_f32_bf16<<<(n + 255) / 256, 256, 0, stream>>>(src, dst, n);
  };
  castw(w_qkv_s, Wqs, 512 * 1536);
  castw(w_qkv_b, Wqb, 512 * 1536);
  castw(w_f1, Wf1, 1024 * 1024);
  castw(w_f2, Wf2, 1024 * 512);
  castw(w_os, Wos, 512 * 512);
  castw(w_ob, Wob, 512 * 512);

  // sem stream: LN -> qkv GEMM (scatter by inverse into sorted domain)
  ln_cast_kernel<<<NTOK / 8, 256, 0, stream>>>(sem, g_sem, b_sem, XLN);
  gemm_bf16_kernel<true><<<dim3(1536 / BN, NTOK / BM), 256, 0, stream>>>(
      XLN, 512, Wqs, 1536, b_qkv_s, QKVS, 1536, NTOK, 1536, 512, 0, inverse, nullptr, 0);
  // bnd stream
  ln_cast_kernel<<<NTOK / 8, 256, 0, stream>>>(bnd, g_bnd, b_bnd, XLN);
  gemm_bf16_kernel<true><<<dim3(1536 / BN, NTOK / BM), 256, 0, stream>>>(
      XLN, 512, Wqb, 1536, b_qkv_b, QKVB, 1536, NTOK, 1536, 512, 0, inverse, nullptr, 0);

  // fusion MLP on concatenated queries
  concat_q_kernel<<<(NTOK * 512) / 256, 256, 0, stream>>>(QKVS, QKVB, QC);
  gemm_bf16_kernel<true><<<dim3(1024 / BN, NTOK / BM), 256, 0, stream>>>(
      QC, 1024, Wf1, 1024, b_f1, H1, 1024, NTOK, 1024, 1024, 1, nullptr, nullptr, 0);
  gemm_bf16_kernel<true><<<dim3(512 / BN, NTOK / BM), 256, 0, stream>>>(
      H1, 1024, Wf2, 512, b_f2, QF, 512, NTOK, 512, 1024, 0, nullptr, nullptr, 0);

  // patch attention (both streams, sorted domain)
  __bf16* ATTNS = QC;
  __bf16* ATTNB = QC + N * 512;
  attn_kernel<<<dim3(PNUM, HNUM / 2), 64, 0, stream>>>(QF, QKVS, ATTNS);
  attn_kernel<<<dim3(PNUM, HNUM / 2), 64, 0, stream>>>(QF, QKVB, ATTNB);

  // output projections: scatter by order back to original domain + residual add
  float* outS = (float*)d_out;
  float* outB = outS + N * CDIM;
  gemm_bf16_kernel<false><<<dim3(512 / BN, NTOK / BM), 256, 0, stream>>>(
      ATTNS, 512, Wos, 512, b_os, outS, 512, NTOK, 512, 512, 0, order, sem, 512);
  gemm_bf16_kernel<false><<<dim3(512 / BN, NTOK / BM), 256, 0, stream>>>(
      ATTNB, 512, Wob, 512, b_ob, outB, 512, NTOK, 512, 512, 0, order, bnd, 512);
}
